// SeAugRumorGNN_33706903339486
// MI455X (gfx1250) — matrix operand order repaired and checked by reading, not compiled, not run on hardware
//
#include <hip/hip_runtime.h>
#include <hip/hip_bf16.h>

// ---------------- problem constants (match reference) ----------------
constexpr int N    = 100000;
constexpr int E    = 1600000;
constexpr int G    = 1024;
constexpr int BASE = 1000;
constexpr int AUG  = 384;
constexpr int H    = 256;
constexpr int C    = 2;
constexpr float EPS = 1e-5f;
constexpr int KFUSE = BASE + AUG;   // 1384

// ---------------- WMMA types ----------------
typedef __attribute__((ext_vector_type(16))) __bf16 v16bf;
typedef __attribute__((ext_vector_type(8)))  float  v8f;

union BfFrag { v16bf v; unsigned u[8]; };

// fp32 -> bf16 (round to nearest even), integer ops only
__device__ __forceinline__ unsigned short f2bf(float f) {
  union { float f; unsigned u; } c; c.f = f;
  unsigned r = c.u + 0x7FFFu + ((c.u >> 16) & 1u);
  return (unsigned short)(r >> 16);
}
__device__ __forceinline__ unsigned pack2(float a, float b) {
  return (unsigned)f2bf(a) | ((unsigned)f2bf(b) << 16);
}

// ---------------- degree / dinv ----------------
__global__ void k_deg_init(float* deg) {
  int i = blockIdx.x * blockDim.x + threadIdx.x;
  if (i < N) deg[i] = 1.0f;                       // self-loop weight
}
__global__ void k_deg_count(const int* __restrict__ dst, float* deg) {
  int e = blockIdx.x * blockDim.x + threadIdx.x;
  if (e < E) atomicAdd(&deg[dst[e]], 1.0f);
}
__global__ void k_dinv(const float* __restrict__ deg, float* dinv) {
  int i = blockIdx.x * blockDim.x + threadIdx.x;
  if (i < N) dinv[i] = rsqrtf(deg[i]);
}

// ---------------- WMMA GEMM: C[N,H] = A[N,K] @ W[K,H] ----------------
// MODE 0: A = concat(x[N,BASE], x_aug[N,AUG]); store ReLU(val + bias)
// MODE 1: A = A0 with lda = H;                 store raw val
// 64x128 tile / 256-thread block, 8 waves each 16x64, double-buffered LDS.
constexpr int BM = 64, BN = 128, BK = 32;

template <int MODE>
__global__ __launch_bounds__(256) void k_gemm_wmma(
    const float* __restrict__ A0, const float* __restrict__ A1,
    const float* __restrict__ W,  const float* __restrict__ bias,
    float* __restrict__ Cout, int nrows, int K)
{
  __shared__ unsigned short sA[2][BM * BK];   // [row][k], pairs contiguous
  __shared__ unsigned short sB[2][BN * BK];   // [col][k] (transposed), pairs contiguous

  const int tid  = threadIdx.x;
  const int lane = tid & 31;
  const int w    = tid >> 5;               // 8 waves
  const int m0   = (w & 3) * 16;           // wave row offset in tile
  const int n0   = (w >> 2) * 64;          // wave col offset in tile (4x 16-wide)
  const int rowBase = blockIdx.x * BM;
  const int colBase = blockIdx.y * BN;

  v8f acc[4] = {};

  // -- staging register files --
  float4 rA[2];        // A: 512 float4 units / 256 thr = 2 each (row = idx/8, k = 4*(idx%8))
  float  rB[16];       // B: 512 (col, 8k) units / 256 thr = 2 each

  auto loadA = [&](int kt) {
#pragma unroll
    for (int i = 0; i < 2; ++i) {
      const int idx = tid + i * 256;
      const int r = idx >> 3, kq = idx & 7;
      const int gr = rowBase + r, gk = kt * BK + 4 * kq;
      float4 v = make_float4(0.f, 0.f, 0.f, 0.f);
      if (gr < nrows && gk < K) {          // BASE/AUG/H all %4==0: float4 never straddles
        const float* s;
        if (MODE == 0) s = (gk < BASE) ? &A0[gr * BASE + gk] : &A1[gr * AUG + (gk - BASE)];
        else           s = &A0[gr * H + gk];
        v = *(const float4*)s;
        __builtin_prefetch(s + 2 * BK, 0, 1);   // pull tile kt+2 (global_prefetch_b8)
      }
      rA[i] = v;
    }
  };
  auto storeA = [&](int buf) {
#pragma unroll
    for (int i = 0; i < 2; ++i) {
      const int idx = tid + i * 256;
      const int r = idx >> 3, k = (idx & 7) * 4;
      unsigned* d = (unsigned*)&sA[buf][r * BK + k];
      d[0] = pack2(rA[i].x, rA[i].y);
      d[1] = pack2(rA[i].z, rA[i].w);
    }
  };
  auto loadB = [&](int kt) {
#pragma unroll
    for (int i = 0; i < 2; ++i) {
      const int idx = tid + i * 256;
      const int c = idx & 127, kg = idx >> 7;   // kg in 0..3 (8 k's each)
#pragma unroll
      for (int v = 0; v < 8; ++v) {
        const int gk = kt * BK + kg * 8 + v;
        rB[i * 8 + v] = (gk < K) ? W[gk * H + (colBase + c)] : 0.0f;
      }
    }
  };
  auto storeB = [&](int buf) {
#pragma unroll
    for (int i = 0; i < 2; ++i) {
      const int idx = tid + i * 256;
      const int c = idx & 127, kg = idx >> 7;
      unsigned* d = (unsigned*)&sB[buf][c * BK + kg * 8];
#pragma unroll
      for (int v = 0; v < 4; ++v)
        d[v] = pack2(rB[i * 8 + 2 * v], rB[i * 8 + 2 * v + 1]);
    }
  };

  const int nk = (K + BK - 1) / BK;

  // prologue: fill buffer 0
  loadA(0); loadB(0);
  storeA(0); storeB(0);
  __syncthreads();

  for (int kt = 0; kt < nk; ++kt) {
    const int cur = kt & 1, nxt = cur ^ 1;
    const bool more = (kt + 1 < nk);
    if (more) { loadA(kt + 1); loadB(kt + 1); }   // global loads in flight over compute

    // A fragment: 16x32 bf16. lane<16 holds K 0..15, lane>=16 holds K 16..31
    BfFrag af;
    const int ar = m0 + (lane & 15);
    const int ka = (lane < 16) ? 0 : 16;
#pragma unroll
    for (int v = 0; v < 4; ++v)
      af.u[v]     = *(const unsigned*)&sA[cur][ar * BK + ka + 2 * v];
#pragma unroll
    for (int v = 0; v < 4; ++v)
      af.u[4 + v] = *(const unsigned*)&sA[cur][ar * BK + ka + 8 + 2 * v];

    // B fragments: 32x16 bf16. lane<16 holds K 0..15, lane>=16 holds K 16..31
    const int kb = (lane < 16) ? 0 : 16;
#pragma unroll
    for (int t = 0; t < 4; ++t) {
      BfFrag bfr;
      const int bc = n0 + 16 * t + (lane & 15);
#pragma unroll
      for (int v = 0; v < 8; ++v)
        bfr.u[v] = *(const unsigned*)&sB[cur][bc * BK + kb + 2 * v];
      acc[t] = __builtin_amdgcn_wmma_f32_16x16x32_bf16(
          false, af.v, false, bfr.v, (short)0, acc[t], false, false);
    }

    if (more) { storeA(nxt); storeB(nxt); }
    __syncthreads();
  }

  // store: C 16x16 f32 layout: VGPR r -> M = r + (lane<16 ? 0 : 8), N = lane%16
  const int rOff = (lane < 16) ? 0 : 8;
  const int cL   = lane & 15;
#pragma unroll
  for (int t = 0; t < 4; ++t) {
    const int gc = colBase + n0 + 16 * t + cL;
#pragma unroll
    for (int r = 0; r < 8; ++r) {
      const int gr = rowBase + m0 + r + rOff;
      if (gr < nrows) {
        float val = acc[t][r];
        if (MODE == 0) val = fmaxf(val + bias[gc], 0.0f);
        Cout[gr * H + gc] = val;
      }
    }
  }
}

// ---------------- GCN aggregation ----------------
// agg[i][j] = hw[i][j] * dinv[i]^2 + b[j]   (self-loop + bias)
__global__ void k_selfloop(const float* __restrict__ hw, const float* __restrict__ dinv,
                           const float* __restrict__ b, float* __restrict__ agg) {
  int idx = blockIdx.x * blockDim.x + threadIdx.x;
  if (idx >= N * H) return;
  int i = idx >> 8, j = idx & (H - 1);
  float d = dinv[i];
  agg[idx] = hw[idx] * d * d + b[j];
}

// one edge per 64 threads, 4 floats per thread (float4 gather + 4 scalar atomics)
__global__ void k_edge_scatter(const float* __restrict__ hw, const float* __restrict__ dinv,
                               const int* __restrict__ src, const int* __restrict__ dst,
                               float* __restrict__ agg) {
  long long idx = (long long)blockIdx.x * blockDim.x + threadIdx.x;
  int e = (int)(idx >> 6);
  if (e >= E) return;
  int j0 = ((int)idx & 63) * 4;
  int s = src[e], d = dst[e];
  float nrm = dinv[s] * dinv[d];
  const float4 v = *(const float4*)&hw[s * H + j0];
  float* dp = &agg[d * H + j0];
  atomicAdd(dp + 0, v.x * nrm);
  atomicAdd(dp + 1, v.y * nrm);
  atomicAdd(dp + 2, v.z * nrm);
  atomicAdd(dp + 3, v.w * nrm);
}

// ---------------- BatchNorm (training stats) ----------------
__global__ void k_zero(float* p, int n) {
  int i = blockIdx.x * blockDim.x + threadIdx.x;
  if (i < n) p[i] = 0.0f;
}
constexpr int BN_ROWS = 256;
__global__ void k_bn_stats(const float* __restrict__ agg, float* sum, float* sumsq) {
  int col = threadIdx.x;                       // blockDim.x == H
  int r0 = blockIdx.x * BN_ROWS;
  int r1 = min(r0 + BN_ROWS, N);
  float s = 0.0f, q = 0.0f;
  for (int r = r0; r < r1; ++r) {
    float v = agg[r * H + col];
    s += v; q += v * v;
  }
  atomicAdd(&sum[col], s);
  atomicAdd(&sumsq[col], q);
}
__global__ void k_bn_finalize(const float* __restrict__ sum, const float* __restrict__ sumsq,
                              const float* __restrict__ g, const float* __restrict__ be,
                              float* scale, float* shift) {
  int j = threadIdx.x;                         // one block of H threads
  float mean = sum[j] / (float)N;
  float var  = sumsq[j] / (float)N - mean * mean;   // biased variance
  float sc   = g[j] * rsqrtf(var + EPS);
  scale[j] = sc;
  shift[j] = be[j] - mean * sc;
}
__global__ void k_bn_apply_relu(const float* __restrict__ agg, const float* __restrict__ scale,
                                const float* __restrict__ shift, float* __restrict__ h) {
  int idx = blockIdx.x * blockDim.x + threadIdx.x;
  if (idx >= N * H) return;
  int j = idx & (H - 1);
  h[idx] = fmaxf(agg[idx] * scale[j] + shift[j], 0.0f);
}

// ---------------- global mean pool + head ----------------
__global__ void k_pool_count(const int* __restrict__ batch, float* cnt) {
  int i = blockIdx.x * blockDim.x + threadIdx.x;
  if (i < N) atomicAdd(&cnt[batch[i]], 1.0f);
}
__global__ void k_pool_accum(const float* __restrict__ h, const int* __restrict__ batch,
                             float* __restrict__ pooled) {
  int idx = blockIdx.x * blockDim.x + threadIdx.x;
  if (idx >= N * H) return;
  int i = idx >> 8, j = idx & (H - 1);
  atomicAdd(&pooled[batch[i] * H + j], h[idx]);
}
__global__ void k_head(const float* __restrict__ pooled, const float* __restrict__ cnt,
                       const float* __restrict__ fcW, const float* __restrict__ fcb,
                       float* __restrict__ out) {
  int gi = blockIdx.x * blockDim.x + threadIdx.x;
  if (gi >= G) return;
  float inv = 1.0f / fmaxf(cnt[gi], 1.0f);
  float l0 = fcb[0], l1 = fcb[1];
  for (int j = 0; j < H; ++j) {
    float p = pooled[gi * H + j] * inv;
    l0 += p * fcW[j * C + 0];
    l1 += p * fcW[j * C + 1];
  }
  float m = fmaxf(l0, l1);
  float lse = m + logf(expf(l0 - m) + expf(l1 - m));
  out[gi * C + 0] = l0 - lse;
  out[gi * C + 1] = l1 - lse;
}

// ---------------- launch ----------------
static inline size_t alignup(size_t x) { return (x + 255) & ~size_t(255); }

extern "C" void kernel_launch(void* const* d_in, const int* in_sizes, int n_in,
                              void* d_out, int out_size, void* d_ws, size_t ws_size,
                              hipStream_t stream) {
  (void)in_sizes; (void)n_in; (void)out_size; (void)ws_size;

  const float* x      = (const float*)d_in[0];
  const float* x_aug  = (const float*)d_in[1];
  const float* fW     = (const float*)d_in[2];
  const float* fb     = (const float*)d_in[3];
  const float* W0     = (const float*)d_in[4];
  const float* b0     = (const float*)d_in[5];
  const float* g0     = (const float*)d_in[6];
  const float* be0    = (const float*)d_in[7];
  const float* W1     = (const float*)d_in[8];
  const float* b1     = (const float*)d_in[9];
  const float* g1     = (const float*)d_in[10];
  const float* be1    = (const float*)d_in[11];
  const float* fcW    = (const float*)d_in[12];
  const float* fcb    = (const float*)d_in[13];
  const int*   eidx   = (const int*)d_in[14];
  const int*   batch  = (const int*)d_in[15];
  const int*   src    = eidx;
  const int*   dst    = eidx + E;
  float* out = (float*)d_out;

  // workspace carve (~308 MB): deg,dinv, h, hw, agg, stats, cnt, pooled
  char* p = (char*)d_ws;
  float* deg    = (float*)p; p += alignup(sizeof(float) * N);
  float* dinv   = (float*)p; p += alignup(sizeof(float) * N);
  float* h      = (float*)p; p += alignup(sizeof(float) * (size_t)N * H);
  float* hw     = (float*)p; p += alignup(sizeof(float) * (size_t)N * H);
  float* agg    = (float*)p; p += alignup(sizeof(float) * (size_t)N * H);
  float* sum    = (float*)p; p += alignup(sizeof(float) * H);
  float* sumsq  = (float*)p; p += alignup(sizeof(float) * H);
  float* scale  = (float*)p; p += alignup(sizeof(float) * H);
  float* shift  = (float*)p; p += alignup(sizeof(float) * H);
  float* cnt    = (float*)p; p += alignup(sizeof(float) * G);
  float* pooled = (float*)p; p += alignup(sizeof(float) * (size_t)G * H);

  const int NH = N * H;
  dim3 blk(256);

  // degrees + dinv
  k_deg_init<<<dim3((N + 255) / 256), blk, 0, stream>>>(deg);
  k_deg_count<<<dim3((E + 255) / 256), blk, 0, stream>>>(dst, deg);
  k_dinv<<<dim3((N + 255) / 256), blk, 0, stream>>>(deg, dinv);

  // fusion GEMM + ReLU -> h
  dim3 ggrid((N + BM - 1) / BM, H / BN);
  k_gemm_wmma<0><<<ggrid, blk, 0, stream>>>(x, x_aug, fW, fb, h, N, KFUSE);

  const float* Ws[2]  = {W0, W1};
  const float* bs[2]  = {b0, b1};
  const float* gs[2]  = {g0, g1};
  const float* bes[2] = {be0, be1};

  for (int l = 0; l < 2; ++l) {
    // hw = h @ Wl (WMMA)
    k_gemm_wmma<1><<<ggrid, blk, 0, stream>>>(h, nullptr, Ws[l], nullptr, hw, N, H);
    // agg = hw * dinv^2 + b
    k_selfloop<<<dim3((NH + 255) / 256), blk, 0, stream>>>(hw, dinv, bs[l], agg);
    // edge scatter-add (L2-resident gathers + atomics)
    k_edge_scatter<<<dim3(((long long)E * 64 + 255) / 256), blk, 0, stream>>>(hw, dinv, src, dst, agg);
    // BatchNorm stats (zero accumulators first; sum & sumsq contiguous)
    k_zero<<<dim3(1), dim3(2 * H), 0, stream>>>(sum, 2 * H);
    k_bn_stats<<<dim3((N + BN_ROWS - 1) / BN_ROWS), dim3(H), 0, stream>>>(agg, sum, sumsq);
    k_bn_finalize<<<dim3(1), dim3(H), 0, stream>>>(sum, sumsq, gs[l], bes[l], scale, shift);
    // h = ReLU(BN(agg))
    k_bn_apply_relu<<<dim3((NH + 255) / 256), blk, 0, stream>>>(agg, scale, shift, h);
  }

  // global mean pool (cnt & pooled contiguous in the carve)
  k_zero<<<dim3((G * H + G + 255) / 256), blk, 0, stream>>>(cnt, G + G * H);
  k_pool_count<<<dim3((N + 255) / 256), blk, 0, stream>>>(batch, cnt);
  k_pool_accum<<<dim3((NH + 255) / 256), blk, 0, stream>>>(h, batch, pooled);

  // head: fc + log_softmax
  k_head<<<dim3((G + 255) / 256), blk, 0, stream>>>(pooled, cnt, fcW, fcb, out);
}
